// Generator_34531537060142
// MI455X (gfx1250) — matrix operand order repaired and checked by reading, not compiled, not run on hardware
//
#include <hip/hip_runtime.h>
#include <hip/hip_bf16.h>
#include <math.h>
#include <stdint.h>

// ---------------------------------------------------------------------------
// Shapes (fixed by the reference): B=16, L=1024, H=512, nh=8, d=64
// ---------------------------------------------------------------------------
#define B_    16
#define L_    1024
#define H_    512
#define NH_   8
#define D_    64
#define BL_   (B_*L_)          // 16384
#define TOT_  (BL_*H_)         // 8388608 elements of x
#define EQ_   0.04419417382415922f   // 1/sqrt(512)
#define LN_EPS_ 1e-5f
#define NOISE_SCALE_ 1e-6f

typedef __attribute__((ext_vector_type(16))) _Float16 v16h;
typedef __attribute__((ext_vector_type(8)))  _Float16 v8h;
typedef __attribute__((ext_vector_type(2)))  _Float16 v2h;
typedef __attribute__((ext_vector_type(8)))  float    v8f;

static __device__ __forceinline__ v8f wmma_f16(v16h a, v16h b, v8f c) {
  // D(16x16,f32) = A(16x32,f16) * B(32x16,f16) + C
  return __builtin_amdgcn_wmma_f32_16x16x32_f16(
      /*neg_a=*/false, a, /*neg_b=*/false, b,
      /*c_mod=*/(short)0, c, /*reuse_a=*/false, /*reuse_b=*/false);
}

// A-fragment (16x32 f16, wave32): lane l, row m = l&15.
// lanes 0-15 hold K = k..k+15; lanes 16-31 hold K = k+16..k+31.
// Caller passes k0 = k + 16*(lane>>4).
static __device__ __forceinline__ v16h load_afrag(const _Float16* row, int k0) {
  union { v16h v; v8h h[2]; } u;
  u.h[0] = *(const v8h*)(row + k0);
  u.h[1] = *(const v8h*)(row + k0 + 8);
  return u.v;
}

// gfx1250 async global->LDS copy (ASYNCcnt-tracked), 16B per lane.
static __device__ __forceinline__ void async_load_b128(uint32_t lds_off,
                                                       const void* gaddr) {
  asm volatile("global_load_async_to_lds_b128 %0, %1, off"
               :: "v"(lds_off), "v"((uint64_t)(uintptr_t)gaddr)
               : "memory");
}
static __device__ __forceinline__ void wait_async0() {
  asm volatile("s_wait_asynccnt 0" ::: "memory");
}

// ---------------------------------------------------------------------------
// 0) fp32 -> f16 weight conversion
// ---------------------------------------------------------------------------
__global__ void k_f32_to_f16(const float* __restrict__ src,
                             _Float16* __restrict__ dst, int n) {
  int i = blockIdx.x * blockDim.x + threadIdx.x;
  if (i < n) dst[i] = (_Float16)src[i];
}

// ---------------------------------------------------------------------------
// 1) styles_in / styles_val  [B,H] = w @ (W*eq)^T + b
// ---------------------------------------------------------------------------
__global__ void k_styles(const float* __restrict__ w,
                         const float* __restrict__ Wsi, const float* __restrict__ bsi,
                         const float* __restrict__ Wsv, const float* __restrict__ bsv,
                         float* __restrict__ styles_in, float* __restrict__ styles_val) {
  int t = blockIdx.x * blockDim.x + threadIdx.x;
  if (t >= B_ * H_) return;
  int b = t >> 9, o = t & (H_ - 1);
  const float* wr = w + b * H_;
  const float* wi = Wsi + o * H_;
  const float* wv = Wsv + o * H_;
  float si = 0.f, sv = 0.f;
  for (int i = 0; i < H_; ++i) { float xv = wr[i]; si += xv * wi[i]; sv += xv * wv[i]; }
  styles_in[t]  = si * EQ_ + bsi[o];
  styles_val[t] = sv * EQ_ + bsv[o];
}

// ---------------------------------------------------------------------------
// 2) demod sigmas (stored as reciprocals): 1/sqrt(sum_i style_i^2 * W_oi^2)
// ---------------------------------------------------------------------------
__global__ void k_sigmas(const float* __restrict__ styles_in,
                         const float* __restrict__ styles_val,
                         const float* __restrict__ Wq, const float* __restrict__ Wk,
                         const float* __restrict__ Wv, const float* __restrict__ Wa,
                         const float* __restrict__ Wr,
                         float* __restrict__ isq, float* __restrict__ isk,
                         float* __restrict__ isv, float* __restrict__ isw,
                         float* __restrict__ isr) {
  int t = blockIdx.x * blockDim.x + threadIdx.x;
  if (t >= B_ * H_) return;
  int b = t >> 9, o = t & (H_ - 1);
  const float* si = styles_in + b * H_;
  const float* sv = styles_val + b * H_;
  float aq = 0.f, ak = 0.f, av = 0.f, aw = 0.f, ar = 0.f;
  for (int i = 0; i < H_; ++i) {
    float s2i = si[i] * si[i];
    float s2v = sv[i] * sv[i];
    float q = Wq[o * H_ + i]; aq += s2i * q * q;
    float k = Wk[o * H_ + i]; ak += s2i * k * k;
    float v = Wv[o * H_ + i]; av += s2i * v * v;
    float a = Wa[o * H_ + i]; aw += s2v * a * a;
    float r = Wr[o * H_ + i]; ar += s2v * r * r;
  }
  isq[t] = rsqrtf(aq); isk[t] = rsqrtf(ak); isv[t] = rsqrtf(av);
  isw[t] = rsqrtf(aw); isr[t] = rsqrtf(ar);
}

// ---------------------------------------------------------------------------
// 3) whole-tensor layernorm: 2-stage deterministic reduction of x*style
// ---------------------------------------------------------------------------
__global__ void k_red1(const float* __restrict__ x,
                       const float* __restrict__ styles_in,
                       float* __restrict__ partials) {
  __shared__ float ssum[256], ssq[256];
  int tid = threadIdx.x;
  float s = 0.f, q = 0.f;
  for (int i = blockIdx.x * blockDim.x + tid; i < TOT_; i += gridDim.x * blockDim.x) {
    int b = i >> 19;           // 1024*512 = 2^19 per batch
    int h = i & (H_ - 1);
    float v = x[i] * styles_in[(b << 9) + h];
    s += v; q += v * v;
  }
  ssum[tid] = s; ssq[tid] = q; __syncthreads();
  for (int st = 128; st > 0; st >>= 1) {
    if (tid < st) { ssum[tid] += ssum[tid + st]; ssq[tid] += ssq[tid + st]; }
    __syncthreads();
  }
  if (tid == 0) { partials[blockIdx.x * 2] = ssum[0]; partials[blockIdx.x * 2 + 1] = ssq[0]; }
}

__global__ void k_red2(const float* __restrict__ partials, float* __restrict__ mu_rs,
                       int nparts) {
  __shared__ float ssum[256], ssq[256];
  int tid = threadIdx.x;
  float s = 0.f, q = 0.f;
  for (int i = tid; i < nparts; i += 256) { s += partials[2 * i]; q += partials[2 * i + 1]; }
  ssum[tid] = s; ssq[tid] = q; __syncthreads();
  for (int st = 128; st > 0; st >>= 1) {
    if (tid < st) { ssum[tid] += ssum[tid + st]; ssq[tid] += ssq[tid + st]; }
    __syncthreads();
  }
  if (tid == 0) {
    float inv = 1.0f / (float)TOT_;
    float mu = ssum[0] * inv;
    float var = ssq[0] * inv - mu * mu;
    mu_rs[0] = mu;
    mu_rs[1] = rsqrtf(var + LN_EPS_);
  }
}

// 4) xn = (x*style - mu)*rs  ->  f16, packed 2-at-a-time
__global__ void k_norm(const float* __restrict__ x, const float* __restrict__ styles_in,
                       const float* __restrict__ mu_rs, _Float16* __restrict__ xn) {
  int i = (blockIdx.x * blockDim.x + threadIdx.x) * 2;
  if (i >= TOT_) return;
  float mu = mu_rs[0], rs = mu_rs[1];
  int b = i >> 19, h = i & (H_ - 1);
  float v0 = (x[i]     * styles_in[(b << 9) + h]     - mu) * rs;
  float v1 = (x[i + 1] * styles_in[(b << 9) + h + 1] - mu) * rs;
  v2h o; o.x = (_Float16)v0; o.y = (_Float16)v1;
  *(v2h*)(xn + i) = o;
}

// ---------------------------------------------------------------------------
// 5) fused Q/K/V projection GEMM (NT): one 16x32 output tile per wave
//    (A-fragment reused across 2 N-tiles x 3 weight matrices = 6 WMMAs/k-step)
// ---------------------------------------------------------------------------
__global__ void __launch_bounds__(256) k_proj(
    const _Float16* __restrict__ xn,
    const _Float16* __restrict__ wq, const _Float16* __restrict__ wk,
    const _Float16* __restrict__ wv,
    const float* __restrict__ isq, const float* __restrict__ isk,
    const float* __restrict__ isv, const float* __restrict__ sval,
    _Float16* __restrict__ Qh, _Float16* __restrict__ Kh,
    _Float16* __restrict__ Vt, _Float16* __restrict__ Vrow) {
  int wave = (blockIdx.x * blockDim.x + threadIdx.x) >> 5;
  int lane = threadIdx.x & 31;
  int mt = wave >> 4;                    // 1024 row tiles
  int np = wave & 15;                    // 16 column-pair tiles (32 cols each)
  int hi = lane >> 4, ln = lane & 15;
  int m = mt * 16;
  int n0 = np * 32 + ln;
  const _Float16* arow = xn + (size_t)(m + ln) * H_;
  const _Float16* bq0 = wq + (size_t)n0 * H_;
  const _Float16* bq1 = bq0 + 16 * H_;
  const _Float16* bk0 = wk + (size_t)n0 * H_;
  const _Float16* bk1 = bk0 + 16 * H_;
  const _Float16* bv0 = wv + (size_t)n0 * H_;
  const _Float16* bv1 = bv0 + 16 * H_;
  v8f cq[2] = {{}, {}}, ck[2] = {{}, {}}, cv[2] = {{}, {}};
  for (int k = 0; k < H_; k += 32) {
    __builtin_prefetch(arow + k + 64, 0, 3);     // global_prefetch_b8
    v16h a = load_afrag(arow, k + hi * 16);
    int kb = k + hi * 16;
    cq[0] = wmma_f16(a, *(const v16h*)(bq0 + kb), cq[0]);
    cq[1] = wmma_f16(a, *(const v16h*)(bq1 + kb), cq[1]);
    ck[0] = wmma_f16(a, *(const v16h*)(bk0 + kb), ck[0]);
    ck[1] = wmma_f16(a, *(const v16h*)(bk1 + kb), ck[1]);
    cv[0] = wmma_f16(a, *(const v16h*)(bv0 + kb), cv[0]);
    cv[1] = wmma_f16(a, *(const v16h*)(bv1 + kb), cv[1]);
  }
  int b = m >> 10;
  int ml0 = m & (L_ - 1);
#pragma unroll
  for (int e = 0; e < 2; ++e) {
    int n = n0 + e * 16;
    float fq = isq[b * H_ + n];
    float fk = isk[b * H_ + n];
    float fv = isv[b * H_ + n] * sval[b * H_ + n];
    int h = n >> 6, dc = n & 63;
    _Float16* qb  = Qh + ((size_t)(b * NH_ + h) * L_) * D_ + dc;
    _Float16* kb_ = Kh + ((size_t)(b * NH_ + h) * L_) * D_ + dc;
    _Float16* vtb = Vt + ((size_t)(b * NH_ + h) * D_ + dc) * L_;
    _Float16* vrb = Vrow + (size_t)(b * L_) * H_ + n;
#pragma unroll
    for (int r = 0; r < 8; ++r) {
      int mr = ml0 + r + 8 * hi;         // C layout: M = r + 8*(lane>>4)
      qb[(size_t)mr * D_]  = (_Float16)(cq[e][r] * fq);
      kb_[(size_t)mr * D_] = (_Float16)(ck[e][r] * fk);
      float vvf = cv[e][r] * fv;
      vtb[mr] = (_Float16)vvf;
      vrb[(size_t)mr * H_] = (_Float16)vvf;
    }
  }
}

// ---------------------------------------------------------------------------
// 6) flash attention: 1024 blocks = (b,h,128-query slab); 8 waves x 16 queries.
//    K/V tiles double-buffered in LDS with global_load_async_to_lds_b128:
//    the async unit fetches tile t+1 while the waves run WMMA on tile t.
// ---------------------------------------------------------------------------
__global__ void __launch_bounds__(256) k_attn(
    const _Float16* __restrict__ Qh, const _Float16* __restrict__ Kh,
    const _Float16* __restrict__ Vt, _Float16* __restrict__ wv_out) {
  __shared__ __align__(16) _Float16 Ksh[2][32][64];   // keys x d   (2 x 4 KB)
  __shared__ __align__(16) _Float16 Vsh[2][64][32];   // d x keys   (2 x 4 KB)
  __shared__ __align__(32) _Float16 Psh[8][16][32];   // per-wave P (8 KB)
  const int KBUF = 32 * 64 * (int)sizeof(_Float16);   // 4096 B per buffer
  int bh = blockIdx.x >> 3;            // 0..127 = b*8+h
  int qblk = blockIdx.x & 7;
  int t = threadIdx.x;
  int w = t >> 5;
  int lane = t & 31;
  int hi = lane >> 4, ln = lane & 15;
  int q0 = qblk * 128 + w * 16;

  const _Float16* qrow = Qh + (size_t)(bh * L_ + q0 + ln) * D_;
  v16h aQ0 = load_afrag(qrow, hi * 16);        // d 0..31
  v16h aQ1 = load_afrag(qrow, 32 + hi * 16);   // d 32..63

  // per-thread source/dest for the cooperative K/V tile staging (16B each)
  int krow = t >> 3, kseg = t & 7;            // 32 rows x 8 segs of 16B
  int vrow = t >> 2, vseg = t & 3;            // 64 rows x 4 segs of 16B
  const _Float16* kgsrc = Kh + (size_t)(bh * L_ + krow) * D_ + kseg * 8;
  const _Float16* vgsrc = Vt + (size_t)(bh * D_ + vrow) * L_ + vseg * 8;
  uint32_t klds0 = (uint32_t)(uintptr_t)&Ksh[0][krow][kseg * 8];
  uint32_t vlds0 = (uint32_t)(uintptr_t)&Vsh[0][vrow][vseg * 8];

  float mrow[8], lrow[8];
  v8f acc[4];
  v8f zero = {};
#pragma unroll
  for (int r = 0; r < 8; ++r) { mrow[r] = -1e30f; lrow[r] = 0.f; }
#pragma unroll
  for (int j = 0; j < 4; ++j) acc[j] = zero;

  // prologue: stage tile 0 into buffer 0
  async_load_b128(klds0, kgsrc);
  async_load_b128(vlds0, vgsrc);

  for (int kt = 0; kt < L_; kt += 32) {
    int cur = (kt >> 5) & 1;
    // current buffer ready; all waves done reading the other buffer
    wait_async0();
    __syncthreads();
    // kick off DMA of the next tile into the other buffer (overlaps compute)
    if (kt + 32 < L_) {
      int nxt = cur ^ 1;
      async_load_b128(klds0 + nxt * KBUF, kgsrc + (size_t)(kt + 32) * D_);
      async_load_b128(vlds0 + nxt * KBUF, vgsrc + (kt + 32));
    }

    // ---- S = Q K^T for this wave's 16 queries x 32 keys ----
    v8f s[2];
#pragma unroll
    for (int ns = 0; ns < 2; ++ns) {
      const _Float16* kr = &Ksh[cur][ns * 16 + ln][0];
      v16h b0 = *(const v16h*)(kr + hi * 16);
      v16h b1 = *(const v16h*)(kr + 32 + hi * 16);
      v8f tacc = zero;
      tacc = wmma_f16(aQ0, b0, tacc);
      tacc = wmma_f16(aQ1, b1, tacc);
      s[ns] = tacc;
    }
    // ---- online softmax over 32 keys; rows live across 16-lane half-groups
#pragma unroll
    for (int r = 0; r < 8; ++r) {
      float s0 = s[0][r], s1 = s[1][r];
      float tm = fmaxf(s0, s1);
      tm = fmaxf(tm, __shfl_xor(tm, 1, 16));
      tm = fmaxf(tm, __shfl_xor(tm, 2, 16));
      tm = fmaxf(tm, __shfl_xor(tm, 4, 16));
      tm = fmaxf(tm, __shfl_xor(tm, 8, 16));
      float mnew = fmaxf(mrow[r], tm);
      float sc = __expf(mrow[r] - mnew);
      float p0 = __expf(s0 - mnew);
      float p1 = __expf(s1 - mnew);
      float ps = p0 + p1;
      ps += __shfl_xor(ps, 1, 16);
      ps += __shfl_xor(ps, 2, 16);
      ps += __shfl_xor(ps, 4, 16);
      ps += __shfl_xor(ps, 8, 16);
      lrow[r] = lrow[r] * sc + ps;
      mrow[r] = mnew;
#pragma unroll
      for (int j = 0; j < 4; ++j) acc[j][r] *= sc;
      Psh[w][r + 8 * hi][ln]      = (_Float16)p0;   // column = key index
      Psh[w][r + 8 * hi][16 + ln] = (_Float16)p1;
    }
    __syncthreads();
    // ---- O += P V ----
    v16h aP = load_afrag(&Psh[w][ln][0], hi * 16);   // P as 16x32 A-fragment
#pragma unroll
    for (int j = 0; j < 4; ++j) {
      const _Float16* vr = &Vsh[cur][j * 16 + ln][0];
      v16h bV = *(const v16h*)(vr + hi * 16);
      acc[j] = wmma_f16(aP, bV, acc[j]);
    }
    // reads of buffer `cur` and Psh are protected by next iteration's barrier
  }
  int b = bh >> 3, h = bh & 7;
#pragma unroll
  for (int j = 0; j < 4; ++j) {
#pragma unroll
    for (int r = 0; r < 8; ++r) {
      int m = q0 + r + 8 * hi;
      float o = acc[j][r] / lrow[r];
      wv_out[(size_t)(b * L_ + m) * H_ + h * D_ + j * 16 + ln] = (_Float16)o;
    }
  }
}

// ---------------------------------------------------------------------------
// 7) fused output: out = wv@Wa^T/sig_w + Vmod@Wr^T/sig_r + bias + noise; leaky
//    one 16x32 output tile per wave (A-fragments reused across 2 N-tiles)
// ---------------------------------------------------------------------------
__global__ void __launch_bounds__(256) k_out(
    const _Float16* __restrict__ wv, const _Float16* __restrict__ vrow_,
    const _Float16* __restrict__ wa, const _Float16* __restrict__ wr,
    const float* __restrict__ isw, const float* __restrict__ isr,
    const float* __restrict__ bias, const float* __restrict__ noise,
    float* __restrict__ out) {
  int wave = (blockIdx.x * blockDim.x + threadIdx.x) >> 5;
  int lane = threadIdx.x & 31;
  int mt = wave >> 4, np = wave & 15;
  int hi = lane >> 4, ln = lane & 15;
  int m = mt * 16;
  int n0 = np * 32 + ln;
  const _Float16* a1 = wv + (size_t)(m + ln) * H_;
  const _Float16* a2 = vrow_ + (size_t)(m + ln) * H_;
  const _Float16* b10 = wa + (size_t)n0 * H_;
  const _Float16* b11 = b10 + 16 * H_;
  const _Float16* b20 = wr + (size_t)n0 * H_;
  const _Float16* b21 = b20 + 16 * H_;
  v8f c1[2] = {{}, {}}, c2[2] = {{}, {}};
  for (int k = 0; k < H_; k += 32) {
    __builtin_prefetch(a1 + k + 64, 0, 3);
    v16h fa1 = load_afrag(a1, k + hi * 16);
    v16h fa2 = load_afrag(a2, k + hi * 16);
    int kb = k + hi * 16;
    c1[0] = wmma_f16(fa1, *(const v16h*)(b10 + kb), c1[0]);
    c1[1] = wmma_f16(fa1, *(const v16h*)(b11 + kb), c1[1]);
    c2[0] = wmma_f16(fa2, *(const v16h*)(b20 + kb), c2[0]);
    c2[1] = wmma_f16(fa2, *(const v16h*)(b21 + kb), c2[1]);
  }
  int b = m >> 10;
#pragma unroll
  for (int e = 0; e < 2; ++e) {
    int n = n0 + e * 16;
    float fw = isw[b * H_ + n], fr = isr[b * H_ + n], bs = bias[n];
#pragma unroll
    for (int r = 0; r < 8; ++r) {
      int mr = m + r + 8 * hi;               // global row in [0, B*L)
      float y = c1[e][r] * fw + c2[e][r] * fr + bs + noise[mr] * NOISE_SCALE_;
      out[(size_t)mr * H_ + n] = y > 0.f ? y : 0.2f * y;
    }
  }
}

// ---------------------------------------------------------------------------
// host side
// ---------------------------------------------------------------------------
extern "C" void kernel_launch(void* const* d_in, const int* in_sizes, int n_in,
                              void* d_out, int out_size, void* d_ws, size_t ws_size,
                              hipStream_t stream) {
  const float* x     = (const float*)d_in[0];
  const float* w     = (const float*)d_in[1];
  const float* noise = (const float*)d_in[2];
  const float* W_si  = (const float*)d_in[3];
  const float* b_si  = (const float*)d_in[4];
  const float* W_sv  = (const float*)d_in[5];
  const float* b_sv  = (const float*)d_in[6];
  const float* W_q   = (const float*)d_in[7];
  const float* W_k   = (const float*)d_in[8];
  const float* W_v   = (const float*)d_in[9];
  const float* W_a   = (const float*)d_in[10];
  const float* W_r   = (const float*)d_in[11];
  const float* bias  = (const float*)d_in[12];
  float* out = (float*)d_out;

  char* ws = (char*)d_ws;
  size_t off = 0;
  auto alloc = [&](size_t bytes) { size_t o = off; off = (off + bytes + 255) & ~(size_t)255; return o; };

  const size_t WBYTES = (size_t)H_ * H_ * sizeof(_Float16);     // 512 KB
  const size_t BIGH   = (size_t)BL_ * H_ * sizeof(_Float16);    // 16 MB

  _Float16* wqh = (_Float16*)(ws + alloc(WBYTES));
  _Float16* wkh = (_Float16*)(ws + alloc(WBYTES));
  _Float16* wvh = (_Float16*)(ws + alloc(WBYTES));
  _Float16* wah = (_Float16*)(ws + alloc(WBYTES));
  _Float16* wrh = (_Float16*)(ws + alloc(WBYTES));
  float* styles_in  = (float*)(ws + alloc(B_ * H_ * sizeof(float)));
  float* styles_val = (float*)(ws + alloc(B_ * H_ * sizeof(float)));
  float* isq = (float*)(ws + alloc(B_ * H_ * sizeof(float)));
  float* isk = (float*)(ws + alloc(B_ * H_ * sizeof(float)));
  float* isv = (float*)(ws + alloc(B_ * H_ * sizeof(float)));
  float* isw = (float*)(ws + alloc(B_ * H_ * sizeof(float)));
  float* isr = (float*)(ws + alloc(B_ * H_ * sizeof(float)));
  float* partials = (float*)(ws + alloc(1024 * 2 * sizeof(float)));
  float* mu_rs = (float*)(ws + alloc(256));
  _Float16* xn   = (_Float16*)(ws + alloc(BIGH));   // reused as wv after attn
  _Float16* Qh   = (_Float16*)(ws + alloc(BIGH));
  _Float16* Kh   = (_Float16*)(ws + alloc(BIGH));
  _Float16* Vt   = (_Float16*)(ws + alloc(BIGH));
  _Float16* Vrow = (_Float16*)(ws + alloc(BIGH));
  _Float16* wvb  = xn;  // alias: xn is dead once k_proj has run

  const int NW = H_ * H_;
  k_f32_to_f16<<<(NW + 255) / 256, 256, 0, stream>>>(W_q, wqh, NW);
  k_f32_to_f16<<<(NW + 255) / 256, 256, 0, stream>>>(W_k, wkh, NW);
  k_f32_to_f16<<<(NW + 255) / 256, 256, 0, stream>>>(W_v, wvh, NW);
  k_f32_to_f16<<<(NW + 255) / 256, 256, 0, stream>>>(W_a, wah, NW);
  k_f32_to_f16<<<(NW + 255) / 256, 256, 0, stream>>>(W_r, wrh, NW);

  k_styles<<<(B_ * H_) / 256, 256, 0, stream>>>(w, W_si, b_si, W_sv, b_sv,
                                                styles_in, styles_val);
  k_sigmas<<<(B_ * H_) / 256, 256, 0, stream>>>(styles_in, styles_val,
                                                W_q, W_k, W_v, W_a, W_r,
                                                isq, isk, isv, isw, isr);
  k_red1<<<1024, 256, 0, stream>>>(x, styles_in, partials);
  k_red2<<<1, 256, 0, stream>>>(partials, mu_rs, 1024);
  k_norm<<<(TOT_ / 2 + 255) / 256, 256, 0, stream>>>(x, styles_in, mu_rs, xn);

  // 1024 M-tiles x 16 N-pair-tiles = 16384 waves / 8 per block = 2048 blocks
  k_proj<<<2048, 256, 0, stream>>>(xn, wqh, wkh, wvh, isq, isk, isv, styles_val,
                                   Qh, Kh, Vt, Vrow);
  // 128 (b,h) pairs * 8 query slabs = 1024 blocks
  k_attn<<<1024, 256, 0, stream>>>(Qh, Kh, Vt, wvb);

  k_out<<<2048, 256, 0, stream>>>(wvb, Vrow, wah, wrh, isw, isr, bias, noise, out);
}
